// DualAttention_46059229282849
// MI455X (gfx1250) — matrix operand order repaired and checked by reading, not compiled, not run on hardware
//
#include <hip/hip_runtime.h>
#include <math.h>

typedef __bf16 bf16;
typedef __attribute__((ext_vector_type(16))) __bf16 v16bf;
typedef __attribute__((ext_vector_type(8)))  __bf16 v8bf;
typedef __attribute__((ext_vector_type(8)))  float  v8f;

enum { DIM = 512, SEQ = 1024, BATCH = 32 };

union V16U { v16bf v; v8bf h[2]; };

__device__ inline v16bf make_v16(v8bf lo, v8bf hi) {
    V16U u; u.h[0] = lo; u.h[1] = hi; return u.v;
}

__device__ inline v8f vzero8() {
    v8f z = {0.f,0.f,0.f,0.f,0.f,0.f,0.f,0.f};
    return z;
}

__device__ inline v8f wmma_bf16(v16bf a, v16bf b, v8f c) {
    // D = A(16x32 bf16) x B(32x16 bf16) + C(16x16 f32)
    return __builtin_amdgcn_wmma_f32_16x16x32_bf16(
        false, a, false, b, (short)0, c, false, false);
}

// ---------------------------------------------------------------------------
// Kernel 1: convert W[k][n] f32 -> Wt[n][k] bf16 (B-fragment friendly: per
// output column n the K-run is contiguous).
// ---------------------------------------------------------------------------
__global__ __launch_bounds__(256) void cvt_wt_kernel(
    const float* __restrict__ Wq, const float* __restrict__ Wk,
    const float* __restrict__ Wv,
    bf16* __restrict__ Wtq, bf16* __restrict__ Wtk, bf16* __restrict__ Wtv) {
    const int idx = blockIdx.x * 256 + threadIdx.x;   // 0 .. 512*512-1
    const int k = idx >> 9;
    const int n = idx & 511;
    const float* Ws[3] = {Wq, Wk, Wv};
    bf16* Wt[3] = {Wtq, Wtk, Wtv};
    const int w = blockIdx.y;
    Wt[w][(size_t)n * DIM + k] = (bf16)Ws[w][(size_t)k * DIM + n];
}

// ---------------------------------------------------------------------------
// Kernel 2: fused QKV GEMM + bias + ReLU, f32 features -> bf16 q/k/v.
// Block: 8 waves, 16-row tile, each wave covers 64 output cols for all three
// weight matrices. A tile staged (f32->bf16) in double-buffered padded LDS.
// ---------------------------------------------------------------------------
__global__ __launch_bounds__(256) void qkv_kernel(
    const float* __restrict__ f1, const float* __restrict__ f2,
    const bf16* __restrict__ Wtq, const bf16* __restrict__ Wtk,
    const bf16* __restrict__ Wtv,
    const float* __restrict__ bq, const float* __restrict__ bk,
    const float* __restrict__ bv,
    bf16* __restrict__ q1, bf16* __restrict__ k1, bf16* __restrict__ v1,
    bf16* __restrict__ q2, bf16* __restrict__ k2, bf16* __restrict__ v2) {

    const int which = blockIdx.y;
    const float* F = which ? f2 : f1;
    bf16* Q = which ? q2 : q1;
    bf16* K = which ? k2 : k1;
    bf16* V = which ? v2 : v1;

    const size_t r0 = (size_t)blockIdx.x * 16;      // row tile in [B*S]
    const int warp = threadIdx.x >> 5;
    const int lane = threadIdx.x & 31;
    const int ln = lane & 15, lh = lane >> 4;
    const int n0 = warp * 64;

    // padded LDS: row stride 40 bf16 (80B = 20 banks) -> conflict-free b128
    __shared__ __align__(16) bf16 As[2][16 * 40];

    v8f accQ[4], accK[4], accV[4];
    #pragma unroll
    for (int i = 0; i < 4; ++i) { accQ[i] = vzero8(); accK[i] = vzero8(); accV[i] = vzero8(); }

    const int arow = threadIdx.x >> 4;        // 0..15
    const int acp  = (threadIdx.x & 15) * 2;  // 0,2,..,30

    for (int ks = 0; ks < 16; ++ks) {
        const int k0 = ks * 32;
        // stage 16x32 f32 tile -> bf16 LDS (one float2 per thread)
        float2 fv = *(const float2*)(F + (r0 + arow) * DIM + k0 + acp);
        bf16* dst = &As[ks & 1][arow * 40 + acp];
        dst[0] = (bf16)fv.x;
        dst[1] = (bf16)fv.y;
        if (ks < 15)
            __builtin_prefetch(F + (r0 + arow) * DIM + k0 + 32 + acp, 0, 0);
        __syncthreads();
        // A fragment: lane half lh -> K run lh*16 .. lh*16+15
        const bf16* ar = &As[ks & 1][ln * 40 + lh * 16];
        v16bf a = make_v16(*(const v8bf*)ar, *(const v8bf*)(ar + 8));
        #pragma unroll
        for (int nt = 0; nt < 4; ++nt) {
            const size_t boff = (size_t)(n0 + nt * 16 + ln) * DIM + k0 + lh * 16;
            accQ[nt] = wmma_bf16(a, *(const v16bf*)(Wtq + boff), accQ[nt]);
            accK[nt] = wmma_bf16(a, *(const v16bf*)(Wtk + boff), accK[nt]);
            accV[nt] = wmma_bf16(a, *(const v16bf*)(Wtv + boff), accV[nt]);
        }
    }

    // epilogue: + bias, ReLU, bf16 store (row-major, score-stage friendly)
    #pragma unroll
    for (int nt = 0; nt < 4; ++nt) {
        const int n = n0 + nt * 16 + ln;
        const float bbq = bq[n], bbk = bk[n], bbv = bv[n];
        #pragma unroll
        for (int v = 0; v < 8; ++v) {
            const size_t o = (r0 + v + 8 * lh) * DIM + n;
            Q[o] = (bf16)fmaxf(accQ[nt][v] + bbq, 0.f);
            K[o] = (bf16)fmaxf(accK[nt][v] + bbk, 0.f);
            V[o] = (bf16)fmaxf(accV[nt][v] + bbv, 0.f);
        }
    }
}

// ---------------------------------------------------------------------------
// Kernel 3: s[b][s] = sum_t tanh(k[b,s,:] . q[b,t,:]).
// Block: 16-row s-tile; 8 waves split the 1024 t's (2 groups of 64 each).
// tanh + row-sum fused into the GEMM epilogue, atomic f32 accumulation.
// ---------------------------------------------------------------------------
__global__ __launch_bounds__(256) void score_kernel(
    const bf16* __restrict__ k1, const bf16* __restrict__ q2,
    const bf16* __restrict__ k2, const bf16* __restrict__ q1,
    float* __restrict__ s1, float* __restrict__ s2) {

    const int pair = blockIdx.y;
    const int b  = blockIdx.x >> 6;
    const int st = blockIdx.x & 63;
    const bf16* A  = (pair ? k2 : k1) + (size_t)b * SEQ * DIM;
    const bf16* Bm = (pair ? q1 : q2) + (size_t)b * SEQ * DIM;
    float* Sout    = (pair ? s2 : s1) + (size_t)b * SEQ;

    const int warp = threadIdx.x >> 5;
    const int lane = threadIdx.x & 31;
    const int ln = lane & 15, lh = lane >> 4;
    const size_t s0 = (size_t)st * 16;

    v8f srow = vzero8();

    for (int tg = 0; tg < 2; ++tg) {
        const int n0 = (warp * 2 + tg) * 64;
        v8f acc[4];
        #pragma unroll
        for (int i = 0; i < 4; ++i) acc[i] = vzero8();

        for (int ks = 0; ks < 16; ++ks) {
            const int k0 = ks * 32;
            const bf16* ap = A + (s0 + ln) * DIM + k0 + lh * 16;
            v16bf a = make_v16(*(const v8bf*)ap, *(const v8bf*)(ap + 8));
            #pragma unroll
            for (int nt = 0; nt < 4; ++nt) {
                const bf16* bp = Bm + (size_t)(n0 + nt * 16 + ln) * DIM + k0 + lh * 16;
                acc[nt] = wmma_bf16(a, *(const v16bf*)bp, acc[nt]);
            }
        }
        #pragma unroll
        for (int nt = 0; nt < 4; ++nt)
            #pragma unroll
            for (int v = 0; v < 8; ++v)
                srow[v] += tanhf(acc[nt][v]);
    }

    // D layout: VGPR v, lane half lh -> row M = v + 8*lh; sum over the 16
    // lanes of each half (the N columns), then atomically add partial rowsum.
    #pragma unroll
    for (int v = 0; v < 8; ++v) {
        float x = srow[v];
        #pragma unroll
        for (int off = 1; off < 16; off <<= 1) x += __shfl_xor(x, off, 16);
        if (ln == 0) atomicAdd(&Sout[s0 + v + 8 * lh], x);
    }
}

// ---------------------------------------------------------------------------
// Kernel 4: softmax over S, context = w.v, fused feature mean, LayerNorm.
// One block per (batch, side).
// ---------------------------------------------------------------------------
__device__ inline float block_reduce_sum(float v, float* red) {
    #pragma unroll
    for (int o = 16; o > 0; o >>= 1) v += __shfl_xor(v, o, 32);
    const int warp = threadIdx.x >> 5, lane = threadIdx.x & 31;
    if (lane == 0) red[warp] = v;
    __syncthreads();
    if (warp == 0) {
        float x = (lane < 8) ? red[lane] : 0.f;
        #pragma unroll
        for (int o = 4; o > 0; o >>= 1) x += __shfl_xor(x, o, 32);
        if (lane == 0) red[0] = x;
    }
    __syncthreads();
    const float r = red[0];
    __syncthreads();
    return r;
}

__device__ inline float block_reduce_max(float v, float* red) {
    #pragma unroll
    for (int o = 16; o > 0; o >>= 1) v = fmaxf(v, __shfl_xor(v, o, 32));
    const int warp = threadIdx.x >> 5, lane = threadIdx.x & 31;
    if (lane == 0) red[warp] = v;
    __syncthreads();
    if (warp == 0) {
        float x = (lane < 8) ? red[lane] : -1e30f;
        #pragma unroll
        for (int o = 4; o > 0; o >>= 1) x = fmaxf(x, __shfl_xor(x, o, 32));
        if (lane == 0) red[0] = x;
    }
    __syncthreads();
    const float r = red[0];
    __syncthreads();
    return r;
}

__global__ __launch_bounds__(256) void finalize_kernel(
    const float* __restrict__ f1, const float* __restrict__ f2,
    const bf16* __restrict__ v1, const bf16* __restrict__ v2,
    const float* __restrict__ s1, const float* __restrict__ s2,
    const float* __restrict__ gamma, const float* __restrict__ beta,
    float* __restrict__ out) {

    const int b = blockIdx.x;
    const int which = blockIdx.y;
    const float* F = (which ? f2 : f1) + (size_t)b * SEQ * DIM;
    const bf16*  V = (which ? v2 : v1) + (size_t)b * SEQ * DIM;
    const float* Sc = (which ? s2 : s1) + (size_t)b * SEQ;
    float* O = out + ((size_t)which * BATCH + b) * DIM;
    const int tid = threadIdx.x;

    __shared__ float wsm[SEQ];
    __shared__ float red[32];

    // softmax over the 1024 scores
    float sv[4];
    float m = -1e30f;
    #pragma unroll
    for (int i = 0; i < 4; ++i) { sv[i] = Sc[tid + i * 256]; m = fmaxf(m, sv[i]); }
    m = block_reduce_max(m, red);
    float partial = 0.f;
    #pragma unroll
    for (int i = 0; i < 4; ++i) {
        const float e = __expf(sv[i] - m);
        wsm[tid + i * 256] = e;
        partial += e;
    }
    const float Z = block_reduce_sum(partial, red);
    const float invZ = 1.f / Z;
    __syncthreads();

    // context (w . v) and feature column-mean, fused
    const int d0 = tid, d1 = tid + 256;
    float c0 = 0.f, c1 = 0.f, mn0 = 0.f, mn1 = 0.f;
    for (int s = 0; s < SEQ; ++s) {
        const float ww = wsm[s];
        c0  += ww * (float)V[(size_t)s * DIM + d0];
        c1  += ww * (float)V[(size_t)s * DIM + d1];
        mn0 += F[(size_t)s * DIM + d0];
        mn1 += F[(size_t)s * DIM + d1];
    }
    const float x0 = mn0 * (1.f / 1024.f) + c0 * invZ;
    const float x1 = mn1 * (1.f / 1024.f) + c1 * invZ;

    // LayerNorm over 512 (2 per thread)
    const float mu = block_reduce_sum(x0 + x1, red) * (1.f / 512.f);
    const float dx0 = x0 - mu, dx1 = x1 - mu;
    const float var = block_reduce_sum(dx0 * dx0 + dx1 * dx1, red) * (1.f / 512.f);
    const float inv = rsqrtf(var + 1e-5f);
    O[d0] = dx0 * inv * gamma[d0] + beta[d0];
    O[d1] = dx1 * inv * gamma[d1] + beta[d1];
}

// ---------------------------------------------------------------------------
extern "C" void kernel_launch(void* const* d_in, const int* in_sizes, int n_in,
                              void* d_out, int out_size, void* d_ws, size_t ws_size,
                              hipStream_t stream) {
    const float* f1 = (const float*)d_in[0];
    const float* f2 = (const float*)d_in[1];
    const float* Wq = (const float*)d_in[2];
    const float* bq = (const float*)d_in[3];
    const float* Wk = (const float*)d_in[4];
    const float* bk = (const float*)d_in[5];
    const float* Wv = (const float*)d_in[6];
    const float* bv = (const float*)d_in[7];
    const float* gamma = (const float*)d_in[8];
    const float* beta  = (const float*)d_in[9];
    float* out = (float*)d_out;

    char* ws = (char*)d_ws;
    const size_t WT_BYTES  = (size_t)DIM * DIM * sizeof(bf16);          // 512 KiB
    const size_t S_BYTES   = (size_t)BATCH * SEQ * sizeof(float);       // 128 KiB
    const size_t QKV_BYTES = (size_t)BATCH * SEQ * DIM * sizeof(bf16);  // 32 MiB

    bf16* Wtq = (bf16*)(ws);
    bf16* Wtk = (bf16*)(ws + WT_BYTES);
    bf16* Wtv = (bf16*)(ws + 2 * WT_BYTES);
    float* s1 = (float*)(ws + 3 * WT_BYTES);
    float* s2 = (float*)(ws + 3 * WT_BYTES + S_BYTES);
    char* qkv = ws + ((size_t)2 << 20);   // 2 MiB aligned region
    bf16* q1 = (bf16*)(qkv + 0 * QKV_BYTES);
    bf16* k1 = (bf16*)(qkv + 1 * QKV_BYTES);
    bf16* v1 = (bf16*)(qkv + 2 * QKV_BYTES);
    bf16* q2 = (bf16*)(qkv + 3 * QKV_BYTES);
    bf16* k2 = (bf16*)(qkv + 4 * QKV_BYTES);
    bf16* v2 = (bf16*)(qkv + 5 * QKV_BYTES);

    cvt_wt_kernel<<<dim3(1024, 3), 256, 0, stream>>>(Wq, Wk, Wv, Wtq, Wtk, Wtv);
    hipMemsetAsync(s1, 0, 2 * S_BYTES, stream);
    qkv_kernel<<<dim3((BATCH * SEQ) / 16, 2), 256, 0, stream>>>(
        f1, f2, Wtq, Wtk, Wtv, bq, bk, bv, q1, k1, v1, q2, k2, v2);
    score_kernel<<<dim3(BATCH * (SEQ / 16), 2), 256, 0, stream>>>(
        k1, q2, k2, q1, s1, s2);
    finalize_kernel<<<dim3(BATCH, 2), 256, 0, stream>>>(
        f1, f2, v1, v2, s1, s2, gamma, beta, out);
}
